// SelectiveStateFusion_11416023072916
// MI455X (gfx1250) — matrix operand order repaired and checked by reading, not compiled, not run on hardware
//
#include <hip/hip_runtime.h>
#include <hip/hip_bf16.h>

// ---------------- problem constants ----------------
#define NSTR 3
#define Bc   2
#define Nn   2048
#define Dd   256
#define Ss   16
#define DCc  4
#define DIi  512
#define DTRr 16
#define BN   (Bc * Nn)          // 4096 rows per stream
#define ROWS (NSTR * BN)        // 12288

typedef __bf16 bf16_t;
typedef __attribute__((ext_vector_type(16))) __bf16 v16bf;
typedef __attribute__((ext_vector_type(8)))  __bf16 v8bf;
typedef __attribute__((ext_vector_type(8)))  float  v8f;
typedef __attribute__((ext_vector_type(4)))  int    v4i;

#define AS1 __attribute__((address_space(1)))
#define AS3 __attribute__((address_space(3)))

// ---- CDNA5 async global->LDS path (probe-verified builtin; v4i* params) ----
#if __has_builtin(__builtin_amdgcn_global_load_async_to_lds_b128)
#define HAVE_ASYNC 1
#endif

__device__ __forceinline__ void cp16_g2l(const bf16_t* g, bf16_t* l) {
#if defined(HAVE_ASYNC)
    __builtin_amdgcn_global_load_async_to_lds_b128(
        (AS1 v4i*)g, (AS3 v4i*)l, 0, 0);
#else
    *(v8bf*)l = *(const v8bf*)g;
#endif
}

#if defined(HAVE_ASYNC)
#if __has_builtin(__builtin_amdgcn_s_wait_asynccnt)
#define WAIT_ASYNC(n) __builtin_amdgcn_s_wait_asynccnt(n)
#else
#define WAIT_ASYNC(n) asm volatile("s_wait_asynccnt %0" ::"n"(n) : "memory")
#endif
#else
#define WAIT_ASYNC(n)
#endif

// ---------------- small helpers ----------------
__global__ void k_convert_bf16(const float* __restrict__ src, bf16_t* __restrict__ dst, long n) {
    long i = (long)blockIdx.x * blockDim.x + threadIdx.x;
    long stride = (long)gridDim.x * blockDim.x;
    for (; i < n; i += stride) dst[i] = (bf16_t)src[i];
}

// ---------------- layernorm (one 256-thread block per row) ----------------
__global__ void k_layernorm(const float* __restrict__ e, const float* __restrict__ bl,
                            const float* __restrict__ sp, const float* __restrict__ nw,
                            const float* __restrict__ nb, bf16_t* __restrict__ xn) {
    int row = blockIdx.x;               // 0..ROWS-1
    int s = row / BN, r = row % BN;
    const float* x = (s == 0 ? e : (s == 1 ? bl : sp)) + (long)r * Dd;
    int t = threadIdx.x;                // 0..255 == D
    float v = x[t];
    __shared__ float red[Dd];
    red[t] = v; __syncthreads();
    for (int o = 128; o > 0; o >>= 1) { if (t < o) red[t] += red[t + o]; __syncthreads(); }
    float mean = red[0] / Dd; __syncthreads();
    float dv = v - mean;
    red[t] = dv * dv; __syncthreads();
    for (int o = 128; o > 0; o >>= 1) { if (t < o) red[t] += red[t + o]; __syncthreads(); }
    float var = red[0] / Dd;
    float y = dv * rsqrtf(var + 1e-5f) * nw[s * Dd + t] + nb[s * Dd + t];
    xn[(long)row * Dd + t] = (bf16_t)y;
}

__device__ __forceinline__ v16bf frag16(v8bf lo, v8bf hi) {
    return __builtin_shufflevector(lo, hi, 0,1,2,3,4,5,6,7,8,9,10,11,12,13,14,15);
}

// ============ LDS-staged, async double-buffered WMMA GEMM ============
//   C[M x Ncol] = A[M x K] * W[Ncol x K]^T     (A,W bf16 row-major, C f32)
// Block = 256 threads (8 waves) -> 64 x 128 C tile; wave(2x4) -> 32 x 32.
// K staged in 32-wide slices: As 64x32, Bs 128x32, double buffered (24 KB).
// Async DMA for slice k+1 overlaps WMMA on slice k (ASYNCcnt pipelining).
// Fragment layouts per CDNA5 ISA 7.12.2 (16-bit A 16x32 / B 32x16).
__global__ void __launch_bounds__(256)
k_gemm_lds(const bf16_t* __restrict__ A, const bf16_t* __restrict__ W,
           float* __restrict__ C, int M, int Ncol, int K,
           long sA, long sW, long sC) {
    int batch = blockIdx.z;
    A += (long)batch * sA; W += (long)batch * sW; C += (long)batch * sC;

    __shared__ bf16_t As[2][64 * 32];
    __shared__ bf16_t Bs[2][128 * 32];

    const int t    = threadIdx.x;
    const int lane = t & 31, wid = t >> 5;
    const int wm   = wid >> 2, wn = wid & 3;       // wave grid 2 x 4
    const int hi   = lane >> 4, l15 = lane & 15;
    const int mblk = blockIdx.y * 64, nblk = blockIdx.x * 128;

    // staging assignment: A tile = 256 x 16B, B tile = 512 x 16B
    const int ar = t >> 2,            aco = (t & 3) * 8;
    const int br0 = t >> 2,           bc0 = (t & 3) * 8;
    const int br1 = (t + 256) >> 2,   bc1 = ((t + 256) & 3) * 8;

    v8f acc[2][2];
    #pragma unroll
    for (int i = 0; i < 2; ++i)
        #pragma unroll
        for (int j = 0; j < 2; ++j)
            #pragma unroll
            for (int q = 0; q < 8; ++q) acc[i][j][q] = 0.0f;

    auto stage = [&](int buf, int kt) {
        cp16_g2l(A + (long)(mblk + ar)  * K + kt + aco, &As[buf][ar  * 32 + aco]);
        cp16_g2l(W + (long)(nblk + br0) * K + kt + bc0, &Bs[buf][br0 * 32 + bc0]);
        cp16_g2l(W + (long)(nblk + br1) * K + kt + bc1, &Bs[buf][br1 * 32 + bc1]);
    };

    const int nk = K >> 5;
    stage(0, 0);
    for (int ks = 0; ks < nk; ++ks) {
        const int cur = ks & 1;
        if (ks + 1 < nk) {
            stage(cur ^ 1, (ks + 1) << 5);   // async DMA overlaps compute below
            WAIT_ASYNC(3);                   // previous stage's 3 DMAs complete
        } else {
            WAIT_ASYNC(0);
        }
        __syncthreads();

        v16bf af[2], bg[2];
        #pragma unroll
        for (int i = 0; i < 2; ++i) {
            const bf16_t* ap = &As[cur][(wm * 32 + i * 16 + l15) * 32 + hi * 8];
            af[i] = frag16(*(const v8bf*)ap, *(const v8bf*)(ap + 16));
        }
        #pragma unroll
        for (int j = 0; j < 2; ++j) {
            const bf16_t* bp = &Bs[cur][(wn * 32 + j * 16 + l15) * 32 + hi * 8];
            bg[j] = frag16(*(const v8bf*)bp, *(const v8bf*)(bp + 16));
        }
        #pragma unroll
        for (int i = 0; i < 2; ++i)
            #pragma unroll
            for (int j = 0; j < 2; ++j)
                acc[i][j] = __builtin_amdgcn_wmma_f32_16x16x32_bf16(
                    false, af[i], false, bg[j], (short)0, acc[i][j], false, false);
        __syncthreads();   // safe to overwrite buffer 'cur' next iteration
    }

    // C layout: VGPR r -> row +r+8*hi (lanes 0-15 / 16-31), col = l15
    #pragma unroll
    for (int i = 0; i < 2; ++i)
        #pragma unroll
        for (int j = 0; j < 2; ++j) {
            int col = nblk + wn * 32 + j * 16 + l15;
            #pragma unroll
            for (int r = 0; r < 8; ++r)
                C[(long)(mblk + wm * 32 + i * 16 + hi * 8 + r) * Ncol + col] = acc[i][j][r];
        }
}

// ============ direct WMMA GEMM for skinny Ncol (x_proj, Ncol=48) ============
template <int NT>
__global__ void k_gemm_bf16(const bf16_t* __restrict__ A, const bf16_t* __restrict__ W,
                            float* __restrict__ C, int M, int Ncol, int K,
                            long sA, long sW, long sC) {
    int batch = blockIdx.z;
    A += (long)batch * sA; W += (long)batch * sW; C += (long)batch * sC;

    int wave = blockIdx.x * (blockDim.x >> 5) + (threadIdx.x >> 5);
    int lane = threadIdx.x & 31;
    int ntiles = (Ncol + 16 * NT - 1) / (16 * NT);
    int mtiles = M >> 4;
    if (wave >= mtiles * ntiles) return;
    int mt = wave / ntiles, nt = wave % ntiles;
    int m0 = mt << 4, n0 = nt * 16 * NT;
    int hi = lane >> 4, l15 = lane & 15;

    v8f acc[NT];
    #pragma unroll
    for (int i = 0; i < NT; ++i)
        #pragma unroll
        for (int q = 0; q < 8; ++q) acc[i][q] = 0.0f;

    const bf16_t* arow = A + (long)(m0 + l15) * K;
    const int akb = hi * 8, bkb = hi * 8;

    for (int k0 = 0; k0 < K; k0 += 32) {
        v16bf afrag = frag16(*(const v8bf*)(arow + k0 + akb),
                             *(const v8bf*)(arow + k0 + 16 + akb));
        #pragma unroll
        for (int i = 0; i < NT; ++i) {
            const bf16_t* brow = W + (long)(n0 + i * 16 + l15) * K + k0 + bkb;
            v16bf bfrag = frag16(*(const v8bf*)brow, *(const v8bf*)(brow + 16));
            acc[i] = __builtin_amdgcn_wmma_f32_16x16x32_bf16(
                false, afrag, false, bfrag, (short)0, acc[i], false, false);
        }
    }
    #pragma unroll
    for (int i = 0; i < NT; ++i) {
        int nc = n0 + i * 16 + l15;
        #pragma unroll
        for (int r = 0; r < 8; ++r)
            C[(long)(m0 + hi * 8 + r) * Ncol + nc] = acc[i][r];
    }
}

// ---------------- depthwise causal conv(4) + bias + SiLU -> xa (bf16) ----------------
__global__ void k_conv_silu(const float* __restrict__ xz, const float* __restrict__ conv_w,
                            const float* __restrict__ conv_b, bf16_t* __restrict__ xa) {
    long total = (long)NSTR * BN * DIi;
    long i = (long)blockIdx.x * blockDim.x + threadIdx.x;
    long stride = (long)gridDim.x * blockDim.x;
    for (; i < total; i += stride) {
        int d = (int)(i % DIi);
        long row = i / DIi;
        int s = (int)(row / BN), bn = (int)(row % BN);
        int b = bn / Nn, n = bn % Nn;
        float acc = conv_b[s * DIi + d];
        #pragma unroll
        for (int k = 0; k < DCc; ++k) {
            int nn = n - (DCc - 1) + k;
            if (nn >= 0)
                acc += xz[(((long)s * BN + (long)b * Nn + nn) * (2 * DIi)) + d] *
                       conv_w[((long)s * DIi + d) * DCc + k];
        }
        float sv = acc / (1.0f + __expf(-acc));
        xa[i] = (bf16_t)sv;
    }
}

// ---------------- dt projection (K=16) + softplus ----------------
__global__ void k_dtproj(const float* __restrict__ xdbl, const float* __restrict__ dt_w,
                         const float* __restrict__ dt_b, float* __restrict__ dtp) {
    long total = (long)NSTR * BN * DIi;
    long i = (long)blockIdx.x * blockDim.x + threadIdx.x;
    long stride = (long)gridDim.x * blockDim.x;
    for (; i < total; i += stride) {
        int d = (int)(i % DIi);
        long row = i / DIi;
        int s = (int)(row / BN);
        const float* xr = xdbl + row * (DTRr + 2 * Ss);
        const float* wr = dt_w + ((long)s * DIi + d) * DTRr;
        float acc = dt_b[s * DIi + d];
        #pragma unroll
        for (int r = 0; r < DTRr; ++r) acc += xr[r] * wr[r];
        float sp = (acc > 20.0f) ? acc : log1pf(__expf(acc));
        dtp[i] = sp;
    }
}

// ---------------- selective scan: 2 channels/wave, state j on lane (j + 16*c) ----------------
__global__ void k_scan(const float* __restrict__ xdbl, const float* __restrict__ dtp,
                       const bf16_t* __restrict__ xa, const float* __restrict__ xz,
                       const float* __restrict__ A_log, const float* __restrict__ Dp,
                       bf16_t* __restrict__ ybf) {
    int wave = blockIdx.x * (blockDim.x >> 5) + (threadIdx.x >> 5);
    int lane = threadIdx.x & 31;
    const int npairs = NSTR * Bc * DIi / 2;     // 1536
    if (wave >= npairs) return;
    int ch = wave * 2 + (lane >> 4);
    int j = lane & 15;
    int s = ch / (Bc * DIi);
    int rem = ch % (Bc * DIi);
    int b = rem / DIi, d = rem % DIi;

    float Aj  = -__expf(A_log[((long)s * DIi + d) * Ss + j]);
    float Dpd = Dp[s * DIi + d];
    float h = 0.0f;
    long rowbase = (long)s * BN + (long)b * Nn;

    for (int n = 0; n < Nn; ++n) {
        long row = rowbase + n;
        float dt  = dtp[row * DIi + d];
        float xav = (float)xa[row * DIi + d];
        float Bm  = xdbl[row * (DTRr + 2 * Ss) + DTRr + j];
        float Cm  = xdbl[row * (DTRr + 2 * Ss) + DTRr + Ss + j];
        float dA  = __expf(dt * Aj);
        h = dA * h + dt * Bm * xav;
        float p = h * Cm;
        p += __shfl_xor(p, 1); p += __shfl_xor(p, 2);
        p += __shfl_xor(p, 4); p += __shfl_xor(p, 8);
        if (j == 0) {
            float z  = xz[row * (2 * DIi) + DIi + d];
            float yf = (p + xav * Dpd) * (z / (1.0f + __expf(-z)));
            ybf[row * DIi + d] = (bf16_t)yf;
        }
    }
}

// ---------------- residual + concat -> proc (f32), multi (bf16) ----------------
__global__ void k_combine(const float* __restrict__ e, const float* __restrict__ bl,
                          const float* __restrict__ sp, const float* __restrict__ outp,
                          float* __restrict__ proc, bf16_t* __restrict__ multi) {
    long total = (long)NSTR * BN * Dd;
    long i = (long)blockIdx.x * blockDim.x + threadIdx.x;
    long stride = (long)gridDim.x * blockDim.x;
    for (; i < total; i += stride) {
        int d = (int)(i % Dd);
        long row = i / Dd;
        int s = (int)(row / BN), bn = (int)(row % BN);
        const float* x = (s == 0 ? e : (s == 1 ? bl : sp));
        float v = x[(long)bn * Dd + d] + outp[i];
        proc[i] = v;
        multi[(long)bn * (NSTR * Dd) + s * Dd + d] = (bf16_t)v;
    }
}

// ---------------- mean-pool over N (one block per (b, e3) column) ----------------
__global__ void k_pool(const float* __restrict__ proc, float* __restrict__ pooled) {
    int o = blockIdx.x;                         // 0 .. B*768-1
    int b = o / (NSTR * Dd), e = o % (NSTR * Dd);
    int s = e / Dd, d = e % Dd;
    int t = threadIdx.x;                        // 256
    float acc = 0.0f;
    for (int n = t; n < Nn; n += 256)
        acc += proc[((long)s * BN + (long)b * Nn + n) * Dd + d];
    __shared__ float red[256];
    red[t] = acc; __syncthreads();
    for (int oo = 128; oo > 0; oo >>= 1) { if (t < oo) red[t] += red[t + oo]; __syncthreads(); }
    if (t == 0) pooled[o] = red[0] / (float)Nn;
}

// ---------------- aggregation head: gelu MLP + double softmax -> coeffs[B][3] ----------------
__global__ void k_agg(const float* __restrict__ pooled, const float* __restrict__ w1,
                      const float* __restrict__ b1, const float* __restrict__ w2,
                      const float* __restrict__ b2, const float* __restrict__ temp,
                      float* __restrict__ coeffs) {
    __shared__ float h[Bc][Dd];
    int t = threadIdx.x;                        // 256
    for (int b = 0; b < Bc; ++b) {
        float acc = b1[t];
        for (int e = 0; e < NSTR * Dd; ++e) acc += pooled[b * (NSTR * Dd) + e] * w1[(long)t * (NSTR * Dd) + e];
        float c = 0.7978845608028654f * (acc + 0.044715f * acc * acc * acc);
        h[b][t] = 0.5f * acc * (1.0f + tanhf(c));
    }
    __syncthreads();
    if (t < Bc) {
        int b = t;
        float lg[3];
        for (int k = 0; k < 3; ++k) {
            float a = b2[k];
            for (int d = 0; d < Dd; ++d) a += h[b][d] * w2[k * Dd + d];
            lg[k] = a;
        }
        float mx = fmaxf(lg[0], fmaxf(lg[1], lg[2]));
        float e0 = __expf(lg[0] - mx), e1 = __expf(lg[1] - mx), e2 = __expf(lg[2] - mx);
        float sm = e0 + e1 + e2;
        float c0 = e0 / sm, c1 = e1 / sm, c2 = e2 / sm;
        float ti = 1.0f / (temp[0] + 1e-6f);
        float l0 = c0 * ti, l1 = c1 * ti, l2 = c2 * ti;
        float mx2 = fmaxf(l0, fmaxf(l1, l2));
        float f0 = __expf(l0 - mx2), f1 = __expf(l1 - mx2), f2 = __expf(l2 - mx2);
        float sm2 = f0 + f1 + f2;
        coeffs[b * 3 + 0] = f0 / sm2;
        coeffs[b * 3 + 1] = f1 / sm2;
        coeffs[b * 3 + 2] = f2 / sm2;
    }
}

// ---------------- epilogue: proj result + bias + coeff-weighted streams ----------------
__global__ void k_final(const float* __restrict__ projres, const float* __restrict__ proj_b,
                        const float* __restrict__ proc, const float* __restrict__ coeffs,
                        float* __restrict__ out) {
    long total = (long)BN * Dd;
    long i = (long)blockIdx.x * blockDim.x + threadIdx.x;
    long stride = (long)gridDim.x * blockDim.x;
    for (; i < total; i += stride) {
        int d = (int)(i % Dd);
        long bn = i / Dd;
        int b = (int)(bn / Nn);
        float acc = projres[i] + proj_b[d];
        #pragma unroll
        for (int s = 0; s < NSTR; ++s)
            acc += coeffs[b * 3 + s] * proc[((long)s * BN + bn) * Dd + d];
        out[i] = acc;
    }
}

// ---------------- launch ----------------
extern "C" void kernel_launch(void* const* d_in, const int* in_sizes, int n_in,
                              void* d_out, int out_size, void* d_ws, size_t ws_size,
                              hipStream_t stream) {
    const float* edge   = (const float*)d_in[0];
    const float* blob   = (const float*)d_in[1];
    const float* spec   = (const float*)d_in[2];
    const float* norm_w = (const float*)d_in[3];
    const float* norm_b = (const float*)d_in[4];
    const float* in_w   = (const float*)d_in[5];
    const float* conv_w = (const float*)d_in[6];
    const float* conv_b = (const float*)d_in[7];
    const float* xproj_w= (const float*)d_in[8];
    const float* dt_w   = (const float*)d_in[9];
    const float* dt_b   = (const float*)d_in[10];
    const float* A_log  = (const float*)d_in[11];
    const float* Dp     = (const float*)d_in[12];
    const float* out_w  = (const float*)d_in[13];
    const float* temp   = (const float*)d_in[14];
    const float* agg_w1 = (const float*)d_in[15];
    const float* agg_b1 = (const float*)d_in[16];
    const float* agg_w2 = (const float*)d_in[17];
    const float* agg_b2 = (const float*)d_in[18];
    const float* proj_w = (const float*)d_in[19];
    const float* proj_b = (const float*)d_in[20];
    float* out = (float*)d_out;

    size_t off = 0;
    auto carve = [&](size_t bytes) -> void* {
        void* p = (char*)d_ws + off;
        off += (bytes + 255) & ~(size_t)255;
        return p;
    };
    bf16_t* wbf_in    = (bf16_t*)carve((size_t)NSTR * 2 * DIi * Dd * 2);
    bf16_t* wbf_xproj = (bf16_t*)carve((size_t)NSTR * (DTRr + 2 * Ss) * DIi * 2);
    bf16_t* wbf_out   = (bf16_t*)carve((size_t)NSTR * Dd * DIi * 2);
    bf16_t* wbf_proj  = (bf16_t*)carve((size_t)Dd * NSTR * Dd * 2);
    bf16_t* xn_bf     = (bf16_t*)carve((size_t)ROWS * Dd * 2);
    float*  xz_f      = (float*) carve((size_t)ROWS * 2 * DIi * 4);
    bf16_t* xa_bf     = (bf16_t*)carve((size_t)ROWS * DIi * 2);
    float*  xdbl_f    = (float*) carve((size_t)ROWS * (DTRr + 2 * Ss) * 4);
    float*  dtp_f     = (float*) carve((size_t)ROWS * DIi * 4);
    bf16_t* y_bf      = (bf16_t*)carve((size_t)ROWS * DIi * 2);
    float*  outp_f    = (float*) carve((size_t)ROWS * Dd * 4);
    float*  proc_f    = (float*) carve((size_t)ROWS * Dd * 4);
    bf16_t* multi_bf  = (bf16_t*)carve((size_t)BN * NSTR * Dd * 2);
    float*  pooled_f  = (float*) carve((size_t)Bc * NSTR * Dd * 4);
    float*  coeffs_f  = (float*) carve((size_t)Bc * 3 * 4);
    float*  projres_f = (float*) carve((size_t)BN * Dd * 4);
    (void)ws_size; (void)in_sizes; (void)n_in; (void)out_size;

    auto cdiv = [](long a, long b) { return (int)((a + b - 1) / b); };

    // 1) weights -> bf16
    k_convert_bf16<<<cdiv((long)NSTR * 2 * DIi * Dd, 256), 256, 0, stream>>>(in_w,   wbf_in,    (long)NSTR * 2 * DIi * Dd);
    k_convert_bf16<<<cdiv((long)NSTR * 48 * DIi,    256), 256, 0, stream>>>(xproj_w,wbf_xproj, (long)NSTR * 48 * DIi);
    k_convert_bf16<<<cdiv((long)NSTR * Dd * DIi,    256), 256, 0, stream>>>(out_w,  wbf_out,   (long)NSTR * Dd * DIi);
    k_convert_bf16<<<cdiv((long)Dd * NSTR * Dd,     256), 256, 0, stream>>>(proj_w, wbf_proj,  (long)Dd * NSTR * Dd);

    // 2) layernorm -> bf16
    k_layernorm<<<ROWS, Dd, 0, stream>>>(edge, blob, spec, norm_w, norm_b, xn_bf);

    // 3) in_proj GEMM [4096x256]x[1024x256]^T per stream  (LDS+async WMMA)
    k_gemm_lds<<<dim3((2 * DIi) / 128, BN / 64, NSTR), 256, 0, stream>>>(
        xn_bf, wbf_in, xz_f, BN, 2 * DIi, Dd,
        (long)BN * Dd, (long)2 * DIi * Dd, (long)BN * 2 * DIi);

    // 4) causal conv + SiLU
    k_conv_silu<<<cdiv((long)ROWS * DIi, 256), 256, 0, stream>>>(xz_f, conv_w, conv_b, xa_bf);

    // 5) x_proj GEMM [4096x512]x[48x512]^T per stream  (direct WMMA, Ncol=48)
    {
        int waves = (BN / 16) * (48 / 16);
        k_gemm_bf16<1><<<dim3(cdiv(waves, 8), 1, NSTR), 256, 0, stream>>>(
            xa_bf, wbf_xproj, xdbl_f, BN, 48, DIi,
            (long)BN * DIi, (long)48 * DIi, (long)BN * 48);
    }

    // 6) dt projection + softplus
    k_dtproj<<<cdiv((long)ROWS * DIi, 256), 256, 0, stream>>>(xdbl_f, dt_w, dt_b, dtp_f);

    // 7) selective scan (fused gate) -> y bf16
    {
        int npairs = NSTR * Bc * DIi / 2;   // 1536 waves
        k_scan<<<cdiv(npairs, 8), 256, 0, stream>>>(xdbl_f, dtp_f, xa_bf, xz_f, A_log, Dp, y_bf);
    }

    // 8) out_proj GEMM [4096x512]x[256x512]^T per stream  (LDS+async WMMA)
    k_gemm_lds<<<dim3(Dd / 128, BN / 64, NSTR), 256, 0, stream>>>(
        y_bf, wbf_out, outp_f, BN, Dd, DIi,
        (long)BN * DIi, (long)Dd * DIi, (long)BN * Dd);

    // 9) residual + concat
    k_combine<<<cdiv((long)ROWS * Dd, 256), 256, 0, stream>>>(edge, blob, spec, outp_f, proc_f, multi_bf);

    // 10) pool + aggregation coeffs
    k_pool<<<Bc * NSTR * Dd, 256, 0, stream>>>(proc_f, pooled_f);
    k_agg<<<1, 256, 0, stream>>>(pooled_f, agg_w1, agg_b1, agg_w2, agg_b2, temp, coeffs_f);

    // 11) final projection GEMM [4096x768]x[256x768]^T  (LDS+async WMMA)
    k_gemm_lds<<<dim3(Dd / 128, BN / 64, 1), 256, 0, stream>>>(
        multi_bf, wbf_proj, projres_f, BN, Dd, NSTR * Dd,
        0L, 0L, 0L);

    // 12) epilogue
    k_final<<<cdiv((long)BN * Dd, 256), 256, 0, stream>>>(projres_f, proj_b, proc_f, coeffs_f, out);
}
